// CrossAttnBlock_10428180595220
// MI455X (gfx1250) — matrix operand order repaired and checked by reading, not compiled, not run on hardware
//
#include <hip/hip_runtime.h>

typedef _Float16 v4h  __attribute__((ext_vector_type(4)));
typedef _Float16 v8h  __attribute__((ext_vector_type(8)));
typedef _Float16 v16h __attribute__((ext_vector_type(16)));
typedef float    v8f  __attribute__((ext_vector_type(8)));

#define B_   4
#define LQ_  2048
#define LK_  2048
#define H_   16
#define DH_  64
#define D_   1024   /* DQ == DC == INNER */

static __device__ __forceinline__ v16h comb16(v8h lo, v8h hi) {
  return __builtin_shufflevector(lo, hi, 0,1,2,3,4,5,6,7,8,9,10,11,12,13,14,15);
}
static __device__ __forceinline__ v8f vzero8() {
  v8f z = {0.f,0.f,0.f,0.f,0.f,0.f,0.f,0.f};
  return z;
}

// ---- CDNA5 async copy: global -> LDS, tracked by ASYNCcnt ------------------
// LDS generic addresses carry the byte offset in bits [31:0], so truncating a
// __shared__ pointer yields the LDS address operand the instruction needs.
static __device__ __forceinline__ unsigned lds_off(const void* p) {
  return (unsigned)(size_t)p;
}
static __device__ __forceinline__ void async_ld_b128(unsigned ldsoff, const void* g) {
  asm volatile("global_load_async_to_lds_b128 %0, %1, off"
               :: "v"(ldsoff), "v"(g) : "memory");
}
static __device__ __forceinline__ void wait_async0() {
  asm volatile("s_wait_asynccnt 0x0" ::: "memory");
}

// ---------------------------------------------------------------------------
// fp32 -> f16 weight pre-conversion (one pass per weight matrix).
// ---------------------------------------------------------------------------
__global__ __launch_bounds__(256)
void cvt16_kernel(const float* __restrict__ src, _Float16* __restrict__ dst) {
  const size_t i = ((size_t)blockIdx.x * 256 + threadIdx.x) * 4;
  const float4 f = *(const float4*)(src + i);
  v4h h = {(_Float16)f.x, (_Float16)f.y, (_Float16)f.z, (_Float16)f.w};
  *(v4h*)(dst + i) = h;
}

// ---------------------------------------------------------------------------
// Per-row mean / rsqrt(var+eps) for LayerNorm fusion. One block per row.
// ---------------------------------------------------------------------------
__global__ __launch_bounds__(256)
void rowstats_kernel(const float* __restrict__ x,
                     float* __restrict__ mu, float* __restrict__ rsg) {
  const int row  = blockIdx.x;
  const int t    = threadIdx.x;
  const int lane = t & 31, w = t >> 5;
  const float4 vv = ((const float4*)(x + (size_t)row * D_))[t];
  float s  = vv.x + vv.y + vv.z + vv.w;
  float s2 = vv.x*vv.x + vv.y*vv.y + vv.z*vv.z + vv.w*vv.w;
  #pragma unroll
  for (int m = 1; m < 32; m <<= 1) {
    s  += __shfl_xor(s,  m, 32);
    s2 += __shfl_xor(s2, m, 32);
  }
  __shared__ float w1[8], w2[8];
  if (lane == 0) { w1[w] = s; w2[w] = s2; }
  __syncthreads();
  if (t == 0) {
    float a = 0.f, bsum = 0.f;
    #pragma unroll
    for (int i = 0; i < 8; i++) { a += w1[i]; bsum += w2[i]; }
    const float mean = a * (1.0f / (float)D_);
    const float var  = bsum * (1.0f / (float)D_) - mean * mean;
    mu[row]  = mean;
    rsg[row] = rsqrtf(var + 1e-5f);
  }
}

// ---------------------------------------------------------------------------
// GEMM C[M,1024] = A[M,1024] * W16[1024,1024] via v_wmma_f32_16x16x32_f16.
// Block tile 64x128, 8 waves, each wave 16x64 (4 accumulators). K-step 32.
// W tile staged via global_load_async_to_lds_b128 (pre-converted f16).
// LN_A: A fp32, LayerNorm fused while staging; else A f16 staged async.
// ---------------------------------------------------------------------------
template<bool LN_A, bool OUT_F16>
__global__ __launch_bounds__(256)
void gemm_kernel(const void* __restrict__ Asrc,
                 const float* __restrict__ mu,    const float* __restrict__ rsg,
                 const float* __restrict__ gamma, const float* __restrict__ beta,
                 const _Float16* __restrict__ W16,
                 void* __restrict__ Cdst) {
  __shared__ _Float16 As[64 * 48];   // 64 rows x 32 k (stride 48: 16B-aligned rows)
  __shared__ _Float16 Bs[32 * 136];  // 32 k x 128 n (stride 136)

  const int t    = threadIdx.x;
  const int lane = t & 31, w = t >> 5;
  const int wm   = w & 3,  wn = w >> 2;          // 4 waves in M, 2 in N
  const int rowBase = blockIdx.y * 64;
  const int nBase   = blockIdx.x * 128;

  v8f acc[4];
  #pragma unroll
  for (int j = 0; j < 4; j++) acc[j] = vzero8();

  for (int k0 = 0; k0 < D_; k0 += 32) {
    __syncthreads();
    // ---- stage W tile (32x128 f16) via async-to-LDS DMA ----
    {
      const int row = t >> 3, col = (t & 7) * 16;
      const _Float16* gp = W16 + (size_t)(k0 + row) * D_ + nBase + col;
      async_ld_b128(lds_off(&Bs[row * 136 + col]),     gp);
      async_ld_b128(lds_off(&Bs[row * 136 + col + 8]), gp + 8);
    }
    // ---- stage A tile (64x32) ----
    {
      const int row = t >> 2, col = (t & 3) * 8;
      if (LN_A) {
        const float* A = (const float*)Asrc;
        const float4* p = (const float4*)(A + (size_t)(rowBase + row) * D_ + k0 + col);
        const float4 x0 = p[0], x1 = p[1];
        const float m = mu[rowBase + row], r = rsg[rowBase + row];
        const float f[8] = {x0.x, x0.y, x0.z, x0.w, x1.x, x1.y, x1.z, x1.w};
        v8h hv;
        #pragma unroll
        for (int i = 0; i < 8; i++)
          hv[i] = (_Float16)((f[i] - m) * r * gamma[k0 + col + i] + beta[k0 + col + i]);
        *(v8h*)&As[row * 48 + col] = hv;
      } else {
        const _Float16* A = (const _Float16*)Asrc;
        async_ld_b128(lds_off(&As[row * 48 + col]),
                      A + (size_t)(rowBase + row) * D_ + k0 + col);
      }
    }
    // prefetch next K-step of the weight panel while DMA + WMMA run
    if (k0 + 32 < D_)
      __builtin_prefetch(W16 + (size_t)(k0 + 32 + (t >> 3)) * D_ + nBase + (t & 7) * 16, 0, 1);

    wait_async0();
    __syncthreads();

    // ---- A fragment: lane L -> row wm*16+(L&15); K lo/hi split per half-wave ----
    const int arow = wm * 16 + (lane & 15);
    const int kb   = (lane < 16) ? 0 : 8;
    const v8h alo = *(const v8h*)&As[arow * 48 + kb];
    const v8h ahi = *(const v8h*)&As[arow * 48 + kb + 16];
    const v16h afr = comb16(alo, ahi);

    #pragma unroll
    for (int j = 0; j < 4; j++) {
      const int bcol = wn * 64 + j * 16;     // B fragment: lane = N column, 16 K halves
      v8h blo, bhi;
      #pragma unroll
      for (int e = 0; e < 8; e++) {
        blo[e] = Bs[(kb + e) * 136 + bcol + (lane & 15)];
        bhi[e] = Bs[(kb + 16 + e) * 136 + bcol + (lane & 15)];
      }
      acc[j] = __builtin_amdgcn_wmma_f32_16x16x32_f16(
                 false, afr, false, comb16(blo, bhi), (short)0, acc[j], false, false);
    }
  }

  // ---- epilogue: C layout col = lane&15, rows = vgpr + 8*(lane>=16) ----
  const int rb = rowBase + wm * 16 + ((lane >> 4) << 3);
  const int cb = nBase + wn * 64 + (lane & 15);
  #pragma unroll
  for (int j = 0; j < 4; j++) {
    #pragma unroll
    for (int i = 0; i < 8; i++) {
      const size_t idx = (size_t)(rb + i) * D_ + cb + j * 16;
      if (OUT_F16) ((_Float16*)Cdst)[idx] = (_Float16)acc[j][i];
      else         ((float*)Cdst)[idx]    = acc[j][i];
    }
  }
}

// ---------------------------------------------------------------------------
// L2 normalize each 64-wide head row (in place, f16). One wave per row.
// use_tau: additionally divide by (tau + 1e-6) (folded softmax scale for q).
// ---------------------------------------------------------------------------
__global__ __launch_bounds__(256)
void l2norm_kernel(_Float16* __restrict__ x, const float* __restrict__ tau_ptr,
                   int use_tau) {
  const int t = threadIdx.x;
  const int lane = t & 31, w = t >> 5;
  const size_t row = (size_t)blockIdx.x * 8 + w;
  _Float16* p = x + row * DH_;
  float a = (float)p[lane * 2];
  float b = (float)p[lane * 2 + 1];
  float ss = a * a + b * b;
  #pragma unroll
  for (int m = 1; m < 32; m <<= 1) ss += __shfl_xor(ss, m, 32);
  float n  = fmaxf(sqrtf(ss), 1e-12f);
  float sc = 1.0f / n;
  if (use_tau) sc = sc / (tau_ptr[0] + 1e-6f);
  p[lane * 2]     = (_Float16)(a * sc);
  p[lane * 2 + 1] = (_Float16)(b * sc);
}

// ---------------------------------------------------------------------------
// Flash attention: grid (LQ/64, B*H). 128 threads = 4 waves; each wave owns
// 16 q rows x full DH=64. Streams 64-key tiles (L2-resident on MI455X's
// 192MB L2): V staged row-major via async-to-LDS DMA, K copied row-major so
// the S-GEMM B-fragment is two 16B LDS reads of a K row; P bounced through
// per-wave LDS (C->A relayout) for the P*V WMMAs. Online softmax with
// 16-lane shuffles.
// ---------------------------------------------------------------------------
__global__ __launch_bounds__(128)
void flash_kernel(const _Float16* __restrict__ q, const _Float16* __restrict__ k,
                  const _Float16* __restrict__ v, const unsigned char* __restrict__ maskp,
                  _Float16* __restrict__ ao) {
  __shared__ _Float16 KT[64 * 72];       // [key][d]  (K tile, row-major)
  __shared__ _Float16 Vs[64 * 72];       // [key][d]
  __shared__ _Float16 Ps[4][16 * 72];    // per-wave P tile (C->A relayout)
  __shared__ float    biasS[64];

  const int t    = threadIdx.x;
  const int lane = t & 31, w = t >> 5;
  const int bh = blockIdx.y;
  const int b  = bh / H_, h = bh % H_;
  const int qtile = blockIdx.x * 64;

  // persistent q fragments for this wave's 16 rows (A layout, 2 K-chunks of 32)
  const int qrow = qtile + w * 16 + (lane & 15);
  const _Float16* qp = q + ((size_t)(b * LQ_ + qrow)) * D_ + h * DH_;
  v16h qa[2];
  #pragma unroll
  for (int c = 0; c < 2; c++) {
    const int kb = c * 32 + ((lane < 16) ? 0 : 8);
    const v8h lo = *(const v8h*)(qp + kb);
    const v8h hi = *(const v8h*)(qp + kb + 16);
    qa[c] = comb16(lo, hi);
  }

  v8f o[4];
  #pragma unroll
  for (int j = 0; j < 4; j++) o[j] = vzero8();
  float mrun[8], lrun[8];
  #pragma unroll
  for (int i = 0; i < 8; i++) { mrun[i] = -1e30f; lrun[i] = 0.0f; }

  for (int kt = 0; kt < LK_ / 64; kt++) {
    __syncthreads();
    // ---- stage V (async DMA) + K (row-major copy) + mask bias ----
    #pragma unroll
    for (int it = 0; it < 4; it++) {
      const int idx = t + it * 128;            // 512 16B-chunks per tile
      const int key = idx >> 3, d8 = idx & 7;
      const size_t g = ((size_t)(b * LK_ + kt * 64 + key)) * D_ + h * DH_ + d8 * 8;
      async_ld_b128(lds_off(&Vs[key * 72 + d8 * 8]), v + g);
      const v8h kv = *(const v8h*)(k + g);
      #pragma unroll
      for (int e = 0; e < 8; e++) KT[key * 72 + d8 * 8 + e] = kv[e];
    }
    if (t < 64) biasS[t] = maskp[(size_t)b * LK_ + kt * 64 + t] ? -1e30f : 0.0f;
    if (kt + 1 < LK_ / 64) {   // prefetch next tile while this one computes
      const size_t gn = ((size_t)(b * LK_ + (kt + 1) * 64 + (t >> 3))) * D_
                        + h * DH_ + (t & 7) * 8;
      __builtin_prefetch(k + gn, 0, 1);
      __builtin_prefetch(v + gn, 0, 1);
    }
    wait_async0();
    __syncthreads();

    // ---- S = q @ K^T (16 x 64) ----
    v8f s[4];
    #pragma unroll
    for (int j = 0; j < 4; j++) s[j] = vzero8();
    #pragma unroll
    for (int c = 0; c < 2; c++) {
      #pragma unroll
      for (int j = 0; j < 4; j++) {
        const int kb2 = c * 32 + ((lane < 16) ? 0 : 8);
        const v8h blo = *(const v8h*)&KT[(j * 16 + (lane & 15)) * 72 + kb2];
        const v8h bhi = *(const v8h*)&KT[(j * 16 + (lane & 15)) * 72 + kb2 + 16];
        s[j] = __builtin_amdgcn_wmma_f32_16x16x32_f16(
                 false, qa[c], false, comb16(blo, bhi), (short)0, s[j], false, false);
      }
    }
    #pragma unroll
    for (int j = 0; j < 4; j++) {
      const float bj = biasS[j * 16 + (lane & 15)];
      #pragma unroll
      for (int i = 0; i < 8; i++) s[j][i] += bj;
    }

    // ---- online softmax: row max, rescale, exp, row sum ----
    #pragma unroll
    for (int i = 0; i < 8; i++) {
      float rm = fmaxf(fmaxf(s[0][i], s[1][i]), fmaxf(s[2][i], s[3][i]));
      rm = fmaxf(rm, __shfl_xor(rm, 1, 32));
      rm = fmaxf(rm, __shfl_xor(rm, 2, 32));
      rm = fmaxf(rm, __shfl_xor(rm, 4, 32));
      rm = fmaxf(rm, __shfl_xor(rm, 8, 32));
      const float mnew = fmaxf(mrun[i], rm);
      const float sc   = __expf(mrun[i] - mnew);
      mrun[i] = mnew;
      lrun[i] *= sc;
      #pragma unroll
      for (int j = 0; j < 4; j++) o[j][i] *= sc;
    }
    float rsum[8];
    #pragma unroll
    for (int i = 0; i < 8; i++) rsum[i] = 0.0f;
    #pragma unroll
    for (int j = 0; j < 4; j++) {
      #pragma unroll
      for (int i = 0; i < 8; i++) {
        const float p = __expf(s[j][i] - mrun[i]);
        s[j][i] = p;
        rsum[i] += p;
      }
    }
    #pragma unroll
    for (int i = 0; i < 8; i++) {
      float r = rsum[i];
      r += __shfl_xor(r, 1, 32);
      r += __shfl_xor(r, 2, 32);
      r += __shfl_xor(r, 4, 32);
      r += __shfl_xor(r, 8, 32);
      lrun[i] += r;
    }

    // ---- P (C layout) -> LDS -> A layout, then O += P @ V ----
    const int prow = (lane >> 4) << 3;
    #pragma unroll
    for (int j = 0; j < 4; j++) {
      #pragma unroll
      for (int i = 0; i < 8; i++)
        Ps[w][(i + prow) * 72 + j * 16 + (lane & 15)] = (_Float16)s[j][i];
    }
    #pragma unroll
    for (int c = 0; c < 2; c++) {
      const int kb = c * 32 + ((lane < 16) ? 0 : 8);
      const v8h plo = *(const v8h*)&Ps[w][(lane & 15) * 72 + kb];
      const v8h phi = *(const v8h*)&Ps[w][(lane & 15) * 72 + kb + 16];
      const v16h pa = comb16(plo, phi);
      #pragma unroll
      for (int j = 0; j < 4; j++) {
        v8h vlo, vhi;
        #pragma unroll
        for (int e = 0; e < 8; e++) {
          vlo[e] = Vs[(kb + e) * 72 + j * 16 + (lane & 15)];
          vhi[e] = Vs[(kb + 16 + e) * 72 + j * 16 + (lane & 15)];
        }
        o[j] = __builtin_amdgcn_wmma_f32_16x16x32_f16(
                 false, pa, false, comb16(vlo, vhi), (short)0, o[j], false, false);
      }
    }
  }

  // ---- finalize: divide by l, store f16 attn-out [tok][h*64+d] ----
  const int rb = qtile + w * 16 + ((lane >> 4) << 3);
  #pragma unroll
  for (int i = 0; i < 8; i++) {
    const float inv = 1.0f / lrun[i];
    #pragma unroll
    for (int j = 0; j < 4; j++)
      ao[((size_t)(b * LQ_ + rb + i)) * D_ + h * DH_ + j * 16 + (lane & 15)] =
          (_Float16)(o[j][i] * inv);
  }
}

// ---------------------------------------------------------------------------
// Host-side orchestration (graph-capture safe: launches only).
// d_in order: x, context, key_padding_mask, ln_q_w, ln_q_b, ln_ctx_w,
//             ln_ctx_b, Wq, Wk, Wv, Wo, tau
// ---------------------------------------------------------------------------
extern "C" void kernel_launch(void* const* d_in, const int* in_sizes, int n_in,
                              void* d_out, int out_size, void* d_ws, size_t ws_size,
                              hipStream_t stream) {
  const float* x      = (const float*)d_in[0];
  const float* ctx    = (const float*)d_in[1];
  const unsigned char* mask = (const unsigned char*)d_in[2];
  const float* ln_q_w = (const float*)d_in[3];
  const float* ln_q_b = (const float*)d_in[4];
  const float* ln_c_w = (const float*)d_in[5];
  const float* ln_c_b = (const float*)d_in[6];
  const float* Wq     = (const float*)d_in[7];
  const float* Wk     = (const float*)d_in[8];
  const float* Wv     = (const float*)d_in[9];
  const float* Wo     = (const float*)d_in[10];
  const float* tau    = (const float*)d_in[11];

  const size_t MB = 1024ull * 1024ull;
  char* ws = (char*)d_ws;
  _Float16* q  = (_Float16*)(ws + 0 * MB);    // 16 MB  [8192,1024] f16
  _Float16* kk = (_Float16*)(ws + 16 * MB);   // 16 MB
  _Float16* vv = (_Float16*)(ws + 32 * MB);   // 16 MB
  _Float16* ao = (_Float16*)(ws + 48 * MB);   // 16 MB
  _Float16* Wq16 = (_Float16*)(ws + 64 * MB); // 2 MB each
  _Float16* Wk16 = (_Float16*)(ws + 66 * MB);
  _Float16* Wv16 = (_Float16*)(ws + 68 * MB);
  _Float16* Wo16 = (_Float16*)(ws + 70 * MB);
  float* muX = (float*)(ws + 72 * MB);
  float* rsX = muX + 8192;
  float* muC = rsX + 8192;
  float* rsC = muC + 8192;

  const int Mtok = B_ * LQ_;   // 8192 (== B_*LK_)

  cvt16_kernel<<<1024, 256, 0, stream>>>(Wq, Wq16);
  cvt16_kernel<<<1024, 256, 0, stream>>>(Wk, Wk16);
  cvt16_kernel<<<1024, 256, 0, stream>>>(Wv, Wv16);
  cvt16_kernel<<<1024, 256, 0, stream>>>(Wo, Wo16);

  rowstats_kernel<<<Mtok, 256, 0, stream>>>(x,   muX, rsX);
  rowstats_kernel<<<Mtok, 256, 0, stream>>>(ctx, muC, rsC);

  dim3 gg(D_ / 128, Mtok / 64);  // (8, 128)
  gemm_kernel<true, true><<<gg, 256, 0, stream>>>(x,   muX, rsX, ln_q_w, ln_q_b, Wq16, q);
  gemm_kernel<true, true><<<gg, 256, 0, stream>>>(ctx, muC, rsC, ln_c_w, ln_c_b, Wk16, kk);
  gemm_kernel<true, true><<<gg, 256, 0, stream>>>(ctx, muC, rsC, ln_c_w, ln_c_b, Wv16, vv);

  const int nheadrows = Mtok * H_;  // 131072 rows of 64
  l2norm_kernel<<<nheadrows / 8, 256, 0, stream>>>(q,  tau, 1);
  l2norm_kernel<<<nheadrows / 8, 256, 0, stream>>>(kk, tau, 0);

  flash_kernel<<<dim3(LQ_ / 64, B_ * H_), 128, 0, stream>>>(q, kk, vv, mask, ao);

  gemm_kernel<false, false><<<gg, 256, 0, stream>>>(ao, nullptr, nullptr, nullptr,
                                                    nullptr, Wo16, d_out);
}
